// TemporalPlusConv_30365418783422
// MI455X (gfx1250) — compile-verified, matrix operand order checked by reading, not run on hardware
//
#include <hip/hip_runtime.h>
#include <hip/hip_bf16.h>
#include <math.h>

// ---------------------------------------------------------------------------
// Hetero-GraphSAGE for MI455X (gfx1250): wave32 + WMMA bf16 16x16x32.
// B-matrix staged pre-transposed so both A and B fragments are contiguous
// 8x-bf16 runs in LDS -> ds_load_b128 pairs.
// ---------------------------------------------------------------------------

#define HDIM 128
#define NIP  50000
#define NCON 200000
#define MB   64          // M rows per GEMM block

typedef __attribute__((ext_vector_type(16))) __bf16 v16bf;
typedef __attribute__((ext_vector_type(8)))  float  v8f;

// ---------------- utility kernels ----------------

__global__ void zero_f32_k(float* __restrict__ p, long n) {
  long i = (long)blockIdx.x * blockDim.x + threadIdx.x;
  if (i < n) p[i] = 0.0f;
}

__global__ void lrelu_k(const float* __restrict__ in, float* __restrict__ out, long n) {
  long i = (long)blockIdx.x * blockDim.x + threadIdx.x;
  if (i < n) {
    float v = in[i];
    out[i] = (v >= 0.0f) ? v : 0.01f * v;
  }
}

// One-shot per SAGE call: W[k][n] (f32, row-major) -> Wt[n][k] (bf16).
// 16K elements; cost is noise next to the edge-scatter traffic.
__global__ void w_transpose_bf16_k(const float* __restrict__ W, __bf16* __restrict__ Wt) {
  int i = blockIdx.x * 256 + threadIdx.x;   // i = k*128 + n (coalesced read)
  if (i < HDIM * HDIM) {
    int k = i >> 7, n = i & 127;
    Wt[n * HDIM + k] = (__bf16)W[i];
  }
}

// One wave32 per edge: gather a 128-float row (float4 per lane) and
// atomic-scatter into the destination accumulator; lane0 bumps the degree.
__global__ __launch_bounds__(256)
void scatter_add_k(const float* __restrict__ xsrc,
                   const int* __restrict__ src, const int* __restrict__ dst,
                   float* __restrict__ agg, float* __restrict__ deg, int nE) {
  int edge = blockIdx.x * (blockDim.x >> 5) + (threadIdx.x >> 5);
  int lane = threadIdx.x & 31;
  if (edge >= nE) return;
  int s = src[edge];
  int d = dst[edge];
  const float4* xs = (const float4*)(xsrc + (long)s * HDIM);
  float4 v = xs[lane];
  float* ag = agg + (long)d * HDIM + lane * 4;
  atomicAdd(ag + 0, v.x);
  atomicAdd(ag + 1, v.y);
  atomicAdd(ag + 2, v.z);
  atomicAdd(ag + 3, v.w);
  if (lane == 0) atomicAdd(deg + d, 1.0f);
}

__global__ void div_deg_k(float* __restrict__ agg, const float* __restrict__ deg, long n) {
  long i = (long)blockIdx.x * blockDim.x + threadIdx.x;
  if (i < n) {
    float dg = deg[i >> 7];            // /HDIM
    agg[i] = agg[i] / fmaxf(dg, 1.0f);
  }
}

// ---------------- fused SAGE GEMM + bias + row L2-norm ----------------
// out[m,:] = normalize( agg[m,:] @ Wl + bl + xdst[m,:] @ Wr )
// Block: 256 threads (8 waves). Wave w owns N-tile n0=w*16 and 4 M-tiles.
// LDS: 32KB weight buffer (Wlt -> Wrt -> row norms) + 32KB stage/out buffer.
__global__ __launch_bounds__(256)
void sage_gemm_k(const float* __restrict__ agg, const float* __restrict__ xdst,
                 const __bf16* __restrict__ Wlt, const float* __restrict__ bl,
                 const __bf16* __restrict__ Wrt, float* __restrict__ out,
                 int nDst, int accum) {
  __shared__ __bf16 sW[HDIM * HDIM];   // 32 KB, W transposed: sW[n][k]
  __shared__ float  sBuf[MB * HDIM];   // 32 KB: bf16 A-stage, then f32 out
  __bf16* sA = (__bf16*)sBuf;
  float*  sO = sBuf;

  const int tid     = threadIdx.x;
  const int wave    = tid >> 5;          // 0..7
  const int lane    = tid & 31;
  const int halfsel = lane >> 4;         // 0/1: K half-group per ISA layout
  const int lrow    = lane & 15;         // M (A/C) or N (B) within tile
  const int m0      = blockIdx.x * MB;
  const int n0      = wave << 4;

  v8f acc[4];
#pragma unroll
  for (int mt = 0; mt < 4; ++mt)
#pragma unroll
    for (int i = 0; i < 8; ++i) acc[mt][i] = 0.0f;

  for (int pass = 0; pass < 2; ++pass) {
    const __bf16* Wt = pass ? Wrt : Wlt;
    const float*  Ag = pass ? xdst : agg;

    // Stage pre-converted transposed weights: 32KB as 2048 uint4 copies.
    {
      const uint4* srcW = (const uint4*)Wt;
      uint4* dstW = (uint4*)sW;
      for (int i = tid; i < (HDIM * HDIM) / 8; i += 256) dstW[i] = srcW[i];
    }
    // Stage A rows [MB][HDIM] as bf16 (float4 reads, 4x bf16 stores).
    for (int i = tid; i < (MB * HDIM) / 4; i += 256) {
      int r = i >> 5, c4 = i & 31;
      int g = m0 + r;
      float4 v = make_float4(0.0f, 0.0f, 0.0f, 0.0f);
      if (g < nDst) v = ((const float4*)Ag)[(long)g * (HDIM / 4) + c4];
      __bf16* d = &sA[r * HDIM + c4 * 4];
      d[0] = (__bf16)v.x; d[1] = (__bf16)v.y; d[2] = (__bf16)v.z; d[3] = (__bf16)v.w;
    }
    __syncthreads();

#pragma unroll
    for (int kc = 0; kc < 4; ++kc) {            // K chunks of 32
      // B fragment: column n0+lrow; two contiguous 8x-bf16 runs in sW[n][k].
      v16bf bfrag;
      {
        const __bf16* bp = &sW[(n0 + lrow) * HDIM + (kc << 5) + (halfsel << 3)];
#pragma unroll
        for (int s = 0; s < 8; ++s) bfrag[s] = bp[s];
#pragma unroll
        for (int s = 0; s < 8; ++s) bfrag[8 + s] = bp[16 + s];
      }
#pragma unroll
      for (int mt = 0; mt < 4; ++mt) {
        // A fragment: row mt*16+lrow; same two contiguous runs in sA[m][k].
        v16bf afrag;
        const __bf16* ap = &sA[((mt << 4) + lrow) * HDIM + (kc << 5) + (halfsel << 3)];
#pragma unroll
        for (int s = 0; s < 8; ++s) afrag[s] = ap[s];
#pragma unroll
        for (int s = 0; s < 8; ++s) afrag[8 + s] = ap[16 + s];
        acc[mt] = __builtin_amdgcn_wmma_f32_16x16x32_bf16(
            false, afrag, false, bfrag, (short)0, acc[mt], false, false);
      }
    }
    __syncthreads();   // before restage (pass 1) / before output phase
  }

  // C layout: VGPR r -> row M = r + halfsel*8, col N = lrow. Add bias, spill to LDS.
#pragma unroll
  for (int mt = 0; mt < 4; ++mt) {
#pragma unroll
    for (int r = 0; r < 8; ++r) {
      int row = (mt << 4) + (halfsel << 3) + r;
      int col = n0 + lrow;
      sO[row * HDIM + col] = acc[mt][r] + bl[col];
    }
  }
  __syncthreads();

  // Row L2 norms (weight LDS is free now — reuse it for the 64 norms).
  float* sNrm = (float*)sW;
  if (tid < MB) {
    float s = 0.0f;
#pragma unroll 8
    for (int c = 0; c < HDIM; ++c) {
      float v = sO[tid * HDIM + c];
      s += v * v;
    }
    sNrm[tid] = 1.0f / fmaxf(sqrtf(s), 1e-12f);
  }
  __syncthreads();

  // Vectorized normalize + (optional) accumulate + store.
  for (int i = tid; i < (MB * HDIM) / 4; i += 256) {
    int r = i >> 5, c4 = i & 31;
    int g = m0 + r;
    if (g < nDst) {
      float rn = sNrm[r];
      float4 v = ((const float4*)sO)[r * (HDIM / 4) + c4];
      v.x *= rn; v.y *= rn; v.z *= rn; v.w *= rn;
      float4* op = (float4*)(out + (long)g * HDIM) + c4;
      if (accum) {
        float4 o = *op;
        o.x += v.x; o.y += v.y; o.z += v.z; o.w += v.w;
        *op = o;
      } else {
        *op = v;
      }
    }
  }
}

// ---------------- host-side orchestration ----------------

static void run_sage(const float* xsrc, const float* xdst,
                     const int* src, const int* dst, int nE, int nDst,
                     const float* Wl, const float* bl, const float* Wr,
                     float* agg, float* deg, float* out,
                     __bf16* wltBuf, __bf16* wrtBuf, int accum,
                     hipStream_t st) {
  long nfeat = (long)nDst * HDIM;
  zero_f32_k<<<dim3((unsigned)((nfeat + 255) / 256)), 256, 0, st>>>(agg, nfeat);
  zero_f32_k<<<dim3((unsigned)((nDst + 255) / 256)), 256, 0, st>>>(deg, (long)nDst);
  scatter_add_k<<<dim3((unsigned)((nE + 7) / 8)), 256, 0, st>>>(xsrc, src, dst, agg, deg, nE);
  div_deg_k<<<dim3((unsigned)((nfeat + 255) / 256)), 256, 0, st>>>(agg, deg, nfeat);
  w_transpose_bf16_k<<<dim3((HDIM * HDIM + 255) / 256), 256, 0, st>>>(Wl, wltBuf);
  w_transpose_bf16_k<<<dim3((HDIM * HDIM + 255) / 256), 256, 0, st>>>(Wr, wrtBuf);
  sage_gemm_k<<<dim3((unsigned)((nDst + MB - 1) / MB)), 256, 0, st>>>(
      agg, xdst, wltBuf, bl, wrtBuf, out, nDst, accum);
}

extern "C" void kernel_launch(void* const* d_in, const int* in_sizes, int n_in,
                              void* d_out, int out_size, void* d_ws, size_t ws_size,
                              hipStream_t stream) {
  (void)n_in; (void)out_size; (void)ws_size;

  const float* x_ip      = (const float*)d_in[0];
  const float* x_con     = (const float*)d_in[1];
  const int*   src_ipcon = (const int*)d_in[2];
  const int*   dst_ipcon = (const int*)d_in[3];
  const int*   src_conip = (const int*)d_in[4];
  const int*   dst_conip = (const int*)d_in[5];
  const int*   src_ipip  = (const int*)d_in[6];
  const int*   dst_ipip  = (const int*)d_in[7];
  const int*   src_csrc  = (const int*)d_in[8];
  const int*   dst_csrc  = (const int*)d_in[9];
  const int*   src_cdst  = (const int*)d_in[10];
  const int*   dst_cdst  = (const int*)d_in[11];
  const float* Wl_t      = (const float*)d_in[12];  // [2][3][H][H]
  const float* bl_t      = (const float*)d_in[13];  // [2][3][H]
  const float* Wr_t      = (const float*)d_in[14];  // [2][3][H][H]
  const float* Wl_s      = (const float*)d_in[15];  // [2][2][H][H]
  const float* bl_s      = (const float*)d_in[16];  // [2][2][H]
  const float* Wr_s      = (const float*)d_in[17];  // [2][2][H][H]

  const int E_ipcon = in_sizes[2];
  const int E_conip = in_sizes[4];
  const int E_ipip  = in_sizes[6];
  const int E_csrc  = in_sizes[8];
  const int E_cdst  = in_sizes[10];

  // Workspace carve-out (float units).
  float* ws = (float*)d_ws;
  size_t off = 0;
  const size_t IPF  = (size_t)NIP * HDIM;
  const size_t CONF = (size_t)NCON * HDIM;
  float* ipF0  = ws + off; off += IPF;
  float* ipF1  = ws + off; off += IPF;
  float* ipO   = ws + off; off += IPF;
  float* conF0 = ws + off; off += CONF;
  float* conF1 = ws + off; off += CONF;
  float* conO  = ws + off; off += CONF;
  float* aggIp = ws + off; off += IPF;
  float* aggCon= ws + off; off += CONF;
  float* degIp = ws + off; off += (size_t)NIP;
  float* degCon= ws + off; off += (size_t)NCON;
  __bf16* wltBuf = (__bf16*)(ws + off); off += (size_t)(HDIM * HDIM) / 2;
  __bf16* wrtBuf = (__bf16*)(ws + off); off += (size_t)(HDIM * HDIM) / 2;

  const float* curIp  = x_ip;     // layer-0 temporal reads the inputs in place
  const float* curCon = x_con;
  float* ipTgt[4]  = { ipF0, ipF1, ipF0, (float*)d_out };
  float* conTgt[4] = { conF0, conF1, conF0, (float*)d_out + IPF };
  int phase = 0;

  const size_t HH = (size_t)HDIM * HDIM;

  for (int l = 0; l < 2; ++l) {
    // ---- temporal HeteroConv ----
    const size_t wt = (size_t)l * 3 * HH;
    const size_t bt = (size_t)l * 3 * HDIM;
    run_sage(curIp, curIp, src_ipip, dst_ipip, E_ipip, NIP,
             Wl_t + wt + 0 * HH, bl_t + bt + 0 * HDIM, Wr_t + wt + 0 * HH,
             aggIp, degIp, ipO, wltBuf, wrtBuf, 0, stream);
    run_sage(curCon, curCon, src_csrc, dst_csrc, E_csrc, NCON,
             Wl_t + wt + 1 * HH, bl_t + bt + 1 * HDIM, Wr_t + wt + 1 * HH,
             aggCon, degCon, conO, wltBuf, wrtBuf, 0, stream);
    run_sage(curCon, curCon, src_cdst, dst_cdst, E_cdst, NCON,
             Wl_t + wt + 2 * HH, bl_t + bt + 2 * HDIM, Wr_t + wt + 2 * HH,
             aggCon, degCon, conO, wltBuf, wrtBuf, 1, stream);  // sum edge types
    lrelu_k<<<dim3((unsigned)((IPF + 255) / 256)), 256, 0, stream>>>(ipO, ipTgt[phase], (long)IPF);
    lrelu_k<<<dim3((unsigned)((CONF + 255) / 256)), 256, 0, stream>>>(conO, conTgt[phase], (long)CONF);
    curIp = ipTgt[phase]; curCon = conTgt[phase]; ++phase;

    // ---- spatial HeteroConv ----
    const size_t wsp = (size_t)l * 2 * HH;
    const size_t bsp = (size_t)l * 2 * HDIM;
    run_sage(curIp, curCon, src_ipcon, dst_ipcon, E_ipcon, NCON,
             Wl_s + wsp + 0 * HH, bl_s + bsp + 0 * HDIM, Wr_s + wsp + 0 * HH,
             aggCon, degCon, conO, wltBuf, wrtBuf, 0, stream);
    run_sage(curCon, curIp, src_conip, dst_conip, E_conip, NIP,
             Wl_s + wsp + 1 * HH, bl_s + bsp + 1 * HDIM, Wr_s + wsp + 1 * HH,
             aggIp, degIp, ipO, wltBuf, wrtBuf, 0, stream);
    lrelu_k<<<dim3((unsigned)((IPF + 255) / 256)), 256, 0, stream>>>(ipO, ipTgt[phase], (long)IPF);
    lrelu_k<<<dim3((unsigned)((CONF + 255) / 256)), 256, 0, stream>>>(conO, conTgt[phase], (long)CONF);
    curIp = ipTgt[phase]; curCon = conTgt[phase]; ++phase;
  }
}
